// EVAAttention_47304769798507
// MI455X (gfx1250) — matrix operand order repaired and checked by reading, not compiled, not run on hardware
//
#include <hip/hip_runtime.h>
#include <hip/hip_bf16.h>

// ---------------------------------------------------------------------------
// Problem constants
// ---------------------------------------------------------------------------
#define B_   16
#define N_   577
#define C_   1024
#define H_   16
#define D_   64
#define M_   (B_ * N_)          // 9232 rows
#define NKPAD 608               // keys padded to multiple of 32 for V^T tiles
#define SCALE_ 0.125f
#define EPS_   1e-6f

typedef __bf16 bf16_t;
typedef __attribute__((ext_vector_type(16))) __bf16 v16bf;
typedef __attribute__((ext_vector_type(8)))  float  v8f;

union FragBF { v16bf v; uint4 q[2]; };

__device__ inline v8f zero8() {
    v8f z;
#pragma unroll
    for (int i = 0; i < 8; i++) z[i] = 0.0f;
    return z;
}

__device__ inline v8f wmma_bf16(v16bf a, v16bf b, v8f c) {
    // D = A(16x32) * B(32x16) + C, f32 accumulate
    return __builtin_amdgcn_wmma_f32_16x16x32_bf16(
        /*neg_a=*/false, a, /*neg_b=*/false, b,
        /*c_mod=*/(short)0, c, /*reuse_a=*/false, /*reuse_b=*/false);
}

// A-fragment (16x32 bf16): lane L holds row (L&15); elems 0..7 = K[k0+8*hi ..],
// elems 8..15 = K[k0+16+8*hi ..]  (per CDNA5 ISA 16-bit A layout)
__device__ inline v16bf load_a_frag(const bf16_t* A, int lda, int row, int k0, int lane) {
    const int hi = lane >> 4;
    const bf16_t* p = A + (size_t)row * lda + k0 + 8 * hi;
    FragBF f;
    f.q[0] = *(const uint4*)(p);
    f.q[1] = *(const uint4*)(p + 16);
    return f.v;
}

// B-fragment (32x16 bf16): lane L holds col (L&15); elems 0..7 = K[k0+8*hi ..], elems 8..15 = K[k0+16+8*hi ..]
// Source matrix is stored [col][k], k contiguous.
__device__ inline v16bf load_b_frag(const bf16_t* Bm, int ldb, int col, int k0, int lane) {
    const int hi = lane >> 4;
    const bf16_t* p = Bm + (size_t)col * ldb + k0 + 8 * hi;
    FragBF f;
    f.q[0] = *(const uint4*)(p);
    f.q[1] = *(const uint4*)(p + 16);
    return f.v;
}

// ---------------------------------------------------------------------------
// fp32 -> bf16 cast
// ---------------------------------------------------------------------------
__global__ void cast_bf16_kernel(const float* __restrict__ src,
                                 bf16_t* __restrict__ dst, int n) {
    int i = blockIdx.x * blockDim.x + threadIdx.x;
    if (i < n) dst[i] = (bf16_t)src[i];
}

// zero the padded key columns of V^T (cols 577..607) so PV WMMA sees no junk
__global__ void vt_pad_kernel(bf16_t* __restrict__ vt) {
    const int total = B_ * H_ * D_ * (NKPAD - N_);
    int i = blockIdx.x * blockDim.x + threadIdx.x;
    if (i >= total) return;
    int col  = N_ + (i % (NKPAD - N_));
    int rowi = i / (NKPAD - N_);
    vt[(size_t)rowi * NKPAD + col] = (bf16_t)0.0f;
}

// ---------------------------------------------------------------------------
// QKV GEMM: qkv[m][o] = sum_c x_bf[m][c] * wqkv_bf[o][c] + b_qkv[o]
// Scatter into q(B,H,N,D), k(B,H,N,D), V^T(B,H,D,NKPAD), all bf16.
// Block = 128 thr (4 waves, 2x2), block tile 128x128, wave tile 64x64.
// ---------------------------------------------------------------------------
__global__ __launch_bounds__(128) void qkv_gemm_kernel(
    const bf16_t* __restrict__ X, const bf16_t* __restrict__ W,
    const float* __restrict__ bias,
    bf16_t* __restrict__ qb, bf16_t* __restrict__ kb, bf16_t* __restrict__ vt)
{
    const int lane = threadIdx.x & 31;
    const int wid  = threadIdx.x >> 5;
    const int wm = wid & 1, wn = wid >> 1;
    const int hi = lane >> 4, l15 = lane & 15;
    const int m0 = blockIdx.x * 128 + wm * 64;
    const int n0 = blockIdx.y * 128 + wn * 64;

    v8f acc[4][4];
#pragma unroll
    for (int i = 0; i < 4; i++)
#pragma unroll
        for (int j = 0; j < 4; j++) acc[i][j] = zero8();

    int rowA[4], colB[4];
#pragma unroll
    for (int t = 0; t < 4; t++) {
        rowA[t] = min(m0 + 16 * t + l15, M_ - 1);
        colB[t] = n0 + 16 * t + l15;
    }

#pragma unroll 1
    for (int k0 = 0; k0 < C_; k0 += 32) {
        v16bf a[4], bb[4];
#pragma unroll
        for (int t = 0; t < 4; t++) a[t]  = load_a_frag(X, C_, rowA[t], k0, lane);
#pragma unroll
        for (int t = 0; t < 4; t++) bb[t] = load_b_frag(W, C_, colB[t], k0, lane);
#pragma unroll
        for (int i = 0; i < 4; i++)
#pragma unroll
            for (int j = 0; j < 4; j++)
                acc[i][j] = wmma_bf16(a[i], bb[j], acc[i][j]);
    }

    // store: scatter into q / k / V^T with bias
#pragma unroll
    for (int j = 0; j < 4; j++) {
        const int col = n0 + 16 * j + l15;       // 0..3071
        const float bv = bias[col];
        const int seg = col >> 10;               // 0=q 1=k 2=v
        const int oo  = col & (C_ - 1);
        const int h = oo >> 6, d = oo & 63;
#pragma unroll
        for (int i = 0; i < 4; i++) {
#pragma unroll
            for (int r = 0; r < 8; r++) {
                const int m = m0 + 16 * i + r + 8 * hi;
                if (m >= M_) continue;
                const int b = m / N_, n = m % N_;
                const bf16_t val = (bf16_t)(acc[i][j][r] + bv);
                const int bh = b * H_ + h;
                if (seg == 0)      qb[((size_t)bh * N_ + n) * D_ + d] = val;
                else if (seg == 1) kb[((size_t)bh * N_ + n) * D_ + d] = val;
                else               vt[((size_t)bh * D_ + d) * NKPAD + n] = val;
            }
        }
    }
}

// ---------------------------------------------------------------------------
// RoPE (GPT-J interleaved, 24x24 2D grid), in-place on q and k, tokens 1..576
// ---------------------------------------------------------------------------
__global__ void rope_kernel(bf16_t* __restrict__ qb, bf16_t* __restrict__ kb) {
    const int total = B_ * H_ * (N_ - 1) * (D_ / 2);
    int idx = blockIdx.x * blockDim.x + threadIdx.x;
    if (idx >= total) return;
    const int pair = idx & 31;           // d/2
    const int rest = idx >> 5;
    const int p    = rest % (N_ - 1);    // token-1, 0..575
    const int bh   = rest / (N_ - 1);
    const int a = p / 24, bc = p % 24;
    const int d0 = pair * 2;
    float tt, fi;
    if (d0 < 32) { tt = (float)a  * (16.0f / 24.0f); fi = (float)pair; }
    else         { tt = (float)bc * (16.0f / 24.0f); fi = (float)(pair - 16); }
    const float freq = __expf(-(fi / 16.0f) * 9.210340371976184f); // ln(10000)
    const float ang  = tt * freq;
    float s, c;
    __sincosf(ang, &s, &c);
    const size_t base = ((size_t)bh * N_ + (p + 1)) * D_ + d0;
    {
        float x1 = (float)qb[base], x2 = (float)qb[base + 1];
        qb[base]     = (bf16_t)(x1 * c - x2 * s);
        qb[base + 1] = (bf16_t)(x2 * c + x1 * s);
    }
    {
        float x1 = (float)kb[base], x2 = (float)kb[base + 1];
        kb[base]     = (bf16_t)(x1 * c - x2 * s);
        kb[base + 1] = (bf16_t)(x2 * c + x1 * s);
    }
}

// ---------------------------------------------------------------------------
// Flash attention: 1 wave per (b,h, 32-query tile). 32 keys per step.
// K/V fragments are reused across the two 16-query sub-tiles (halves L2 reads).
// QK: 8 WMMAs / step; P routed C-layout->A-layout via LDS; PV: 8 WMMAs / step.
// ---------------------------------------------------------------------------
__global__ __launch_bounds__(32) void attn_kernel(
    const bf16_t* __restrict__ qb, const bf16_t* __restrict__ kb,
    const bf16_t* __restrict__ vt, float* __restrict__ att)
{
    __shared__ bf16_t Pld[2][16][32];    // [q-subtile][query row][key-in-chunk]
    const int lane = threadIdx.x & 31;
    const int hi = lane >> 4, l15 = lane & 15;
    const int qt = blockIdx.x;           // 0..18
    const int bh = blockIdx.y;           // 0..255
    const int b = bh >> 4, h = bh & 15;
    const int qrow0 = qt * 32;

    const bf16_t* Q = qb + (size_t)bh * N_ * D_;
    const bf16_t* K = kb + (size_t)bh * N_ * D_;
    const bf16_t* V = vt + (size_t)bh * D_ * NKPAD;

    v16bf qf[2][2];
#pragma unroll
    for (int qh = 0; qh < 2; qh++) {
        const int qrow = min(qrow0 + 16 * qh + l15, N_ - 1);
        qf[qh][0] = load_a_frag(Q, D_, qrow, 0,  lane);
        qf[qh][1] = load_a_frag(Q, D_, qrow, 32, lane);
    }

    float mrow[2][8], lrow[2][8];
    v8f acc[2][4];
#pragma unroll
    for (int qh = 0; qh < 2; qh++) {
#pragma unroll
        for (int r = 0; r < 8; r++) { mrow[qh][r] = -1e30f; lrow[qh][r] = 0.0f; }
#pragma unroll
        for (int dt = 0; dt < 4; dt++) acc[qh][dt] = zero8();
    }

#pragma unroll 1
    for (int kc0 = 0; kc0 < N_; kc0 += 32) {
        v8f s[2][2];                     // [q-subtile][key half]
#pragma unroll
        for (int c = 0; c < 2; c++) {
            const int key16 = kc0 + c * 16;
            const int krow  = min(key16 + l15, N_ - 1);
            v16bf kf0 = load_b_frag(K, D_, krow, 0,  lane);
            v16bf kf1 = load_b_frag(K, D_, krow, 32, lane);
            const bool valid = (key16 + l15) < N_;   // this lane's column
#pragma unroll
            for (int qh = 0; qh < 2; qh++) {
                v8f sf = zero8();
                sf = wmma_bf16(qf[qh][0], kf0, sf);
                sf = wmma_bf16(qf[qh][1], kf1, sf);
#pragma unroll
                for (int r = 0; r < 8; r++)
                    sf[r] = valid ? sf[r] * SCALE_ : -1e30f;
                s[qh][c] = sf;
            }
        }

        // online softmax stats per q-subtile
        v16bf pa[2];
        __syncthreads();
#pragma unroll
        for (int qh = 0; qh < 2; qh++) {
            float mnew[8], alpha[8];
#pragma unroll
            for (int r = 0; r < 8; r++) {
                float mc = fmaxf(s[qh][0][r], s[qh][1][r]);
#pragma unroll
                for (int off = 1; off < 16; off <<= 1)
                    mc = fmaxf(mc, __shfl_xor(mc, off, 32));
                mnew[r]      = fmaxf(mrow[qh][r], mc);
                alpha[r]     = __expf(mrow[qh][r] - mnew[r]);
                mrow[qh][r]  = mnew[r];
            }
#pragma unroll
            for (int r = 0; r < 8; r++) {
                const float p0 = __expf(s[qh][0][r] - mnew[r]);
                const float p1 = __expf(s[qh][1][r] - mnew[r]);
                float ps = p0 + p1;
#pragma unroll
                for (int off = 1; off < 16; off <<= 1)
                    ps += __shfl_xor(ps, off, 32);
                lrow[qh][r] = lrow[qh][r] * alpha[r] + ps;
                Pld[qh][r + 8 * hi][l15]      = (bf16_t)p0;
                Pld[qh][r + 8 * hi][16 + l15] = (bf16_t)p1;
            }
#pragma unroll
            for (int dt = 0; dt < 4; dt++)
#pragma unroll
                for (int r = 0; r < 8; r++)
                    acc[qh][dt][r] *= alpha[r];
        }
        __syncthreads();
#pragma unroll
        for (int qh = 0; qh < 2; qh++) {
            FragBF pf;
            pf.q[0] = *(const uint4*)&Pld[qh][l15][8 * hi];
            pf.q[1] = *(const uint4*)&Pld[qh][l15][16 + 8 * hi];
            pa[qh] = pf.v;
        }

        // PV: V^T rows are key-contiguous; each V fragment feeds both q-subtiles
#pragma unroll
        for (int dt = 0; dt < 4; dt++) {
            v16bf vf = load_b_frag(V, NKPAD, dt * 16 + l15, kc0, lane);
            acc[0][dt] = wmma_bf16(pa[0], vf, acc[0][dt]);
            acc[1][dt] = wmma_bf16(pa[1], vf, acc[1][dt]);
        }
    }

    // write O (f32) to att[b][n][h*64+d]
#pragma unroll
    for (int qh = 0; qh < 2; qh++) {
#pragma unroll
        for (int r = 0; r < 8; r++) {
            const int m = qrow0 + 16 * qh + r + 8 * hi;
            if (m >= N_) continue;
            const float inv = 1.0f / lrow[qh][r];
#pragma unroll
            for (int dt = 0; dt < 4; dt++) {
                att[((size_t)(b * N_ + m)) * C_ + h * D_ + dt * 16 + l15] =
                    acc[qh][dt][r] * inv;
            }
        }
    }
}

// ---------------------------------------------------------------------------
// LayerNorm over C=1024, one block per row, output bf16 for the proj GEMM
// ---------------------------------------------------------------------------
__global__ __launch_bounds__(256) void ln_kernel(
    const float* __restrict__ att, const float* __restrict__ g,
    const float* __restrict__ be, bf16_t* __restrict__ xln)
{
    __shared__ float red[16];
    const int row = blockIdx.x;
    const float* x = att + (size_t)row * C_;
    float vals[4], sum = 0.0f, sq = 0.0f;
#pragma unroll
    for (int i = 0; i < 4; i++) {
        vals[i] = x[threadIdx.x + i * 256];
        sum += vals[i];
        sq  += vals[i] * vals[i];
    }
#pragma unroll
    for (int off = 1; off < 32; off <<= 1) {
        sum += __shfl_xor(sum, off, 32);
        sq  += __shfl_xor(sq,  off, 32);
    }
    const int wid = threadIdx.x >> 5;
    if ((threadIdx.x & 31) == 0) { red[wid] = sum; red[8 + wid] = sq; }
    __syncthreads();
    float tsum = 0.0f, tsq = 0.0f;
#pragma unroll
    for (int w = 0; w < 8; w++) { tsum += red[w]; tsq += red[8 + w]; }
    const float mu   = tsum * (1.0f / C_);
    const float var  = tsq * (1.0f / C_) - mu * mu;
    const float rstd = rsqrtf(var + EPS_);
#pragma unroll
    for (int i = 0; i < 4; i++) {
        const int c = threadIdx.x + i * 256;
        const float y = (vals[i] - mu) * rstd * g[c] + be[c];
        xln[(size_t)row * C_ + c] = (bf16_t)y;
    }
}

// ---------------------------------------------------------------------------
// Output projection: out[m][o] = sum_c xln[m][c]*wproj[o][c] + b_proj[o] (f32)
// Block tile 128x128, wave tile 64x64 (same scheme as QKV GEMM).
// ---------------------------------------------------------------------------
__global__ __launch_bounds__(128) void proj_gemm_kernel(
    const bf16_t* __restrict__ X, const bf16_t* __restrict__ W,
    const float* __restrict__ bias, float* __restrict__ out)
{
    const int lane = threadIdx.x & 31;
    const int wid  = threadIdx.x >> 5;
    const int wm = wid & 1, wn = wid >> 1;
    const int hi = lane >> 4, l15 = lane & 15;
    const int m0 = blockIdx.x * 128 + wm * 64;
    const int n0 = blockIdx.y * 128 + wn * 64;

    v8f acc[4][4];
#pragma unroll
    for (int i = 0; i < 4; i++)
#pragma unroll
        for (int j = 0; j < 4; j++) acc[i][j] = zero8();

    int rowA[4], colB[4];
#pragma unroll
    for (int t = 0; t < 4; t++) {
        rowA[t] = min(m0 + 16 * t + l15, M_ - 1);
        colB[t] = n0 + 16 * t + l15;
    }

#pragma unroll 1
    for (int k0 = 0; k0 < C_; k0 += 32) {
        v16bf a[4], bb[4];
#pragma unroll
        for (int t = 0; t < 4; t++) a[t]  = load_a_frag(X, C_, rowA[t], k0, lane);
#pragma unroll
        for (int t = 0; t < 4; t++) bb[t] = load_b_frag(W, C_, colB[t], k0, lane);
#pragma unroll
        for (int i = 0; i < 4; i++)
#pragma unroll
            for (int j = 0; j < 4; j++)
                acc[i][j] = wmma_bf16(a[i], bb[j], acc[i][j]);
    }

#pragma unroll
    for (int j = 0; j < 4; j++) {
        const int col = n0 + 16 * j + l15;
        const float bv = bias[col];
#pragma unroll
        for (int i = 0; i < 4; i++) {
#pragma unroll
            for (int r = 0; r < 8; r++) {
                const int m = m0 + 16 * i + r + 8 * hi;
                if (m < M_)
                    out[(size_t)m * C_ + col] = acc[i][j][r] + bv;
            }
        }
    }
}

// ---------------------------------------------------------------------------
// Host-side launch
// ---------------------------------------------------------------------------
extern "C" void kernel_launch(void* const* d_in, const int* in_sizes, int n_in,
                              void* d_out, int out_size, void* d_ws, size_t ws_size,
                              hipStream_t stream) {
    (void)in_sizes; (void)n_in; (void)out_size; (void)ws_size;
    const float* x      = (const float*)d_in[0];   // (B,N,C)
    const float* w_qkv  = (const float*)d_in[1];   // (3C,C)
    const float* b_qkv  = (const float*)d_in[2];   // (3C)
    const float* ln_g   = (const float*)d_in[3];   // (C)
    const float* ln_b   = (const float*)d_in[4];   // (C)
    const float* w_proj = (const float*)d_in[5];   // (C,C)
    const float* b_proj = (const float*)d_in[6];   // (C)
    float* out = (float*)d_out;

    // workspace layout (256B aligned)
    char* ws = (char*)d_ws;
    size_t off = 0;
    auto take = [&](size_t bytes) -> char* {
        char* p = ws + off;
        off += (bytes + 255) & ~(size_t)255;
        return p;
    };
    bf16_t* xbf   = (bf16_t*)take((size_t)M_ * C_ * 2);
    bf16_t* wqbf  = (bf16_t*)take((size_t)3 * C_ * C_ * 2);
    bf16_t* wpbf  = (bf16_t*)take((size_t)C_ * C_ * 2);
    bf16_t* qb    = (bf16_t*)take((size_t)B_ * H_ * N_ * D_ * 2);
    bf16_t* kb    = (bf16_t*)take((size_t)B_ * H_ * N_ * D_ * 2);
    bf16_t* vt    = (bf16_t*)take((size_t)B_ * H_ * D_ * NKPAD * 2);
    float*  att   = (float*) take((size_t)M_ * C_ * 4);
    bf16_t* xln   = (bf16_t*)take((size_t)M_ * C_ * 2);

    // 1. casts
    {
        int n = M_ * C_;
        cast_bf16_kernel<<<(n + 255) / 256, 256, 0, stream>>>(x, xbf, n);
        n = 3 * C_ * C_;
        cast_bf16_kernel<<<(n + 255) / 256, 256, 0, stream>>>(w_qkv, wqbf, n);
        n = C_ * C_;
        cast_bf16_kernel<<<(n + 255) / 256, 256, 0, stream>>>(w_proj, wpbf, n);
    }
    // 2. zero V^T pad columns
    {
        int n = B_ * H_ * D_ * (NKPAD - N_);
        vt_pad_kernel<<<(n + 255) / 256, 256, 0, stream>>>(vt);
    }
    // 3. QKV GEMM (M=9232, N=3072, K=1024), 128x128 block tiles
    {
        dim3 grid((M_ + 127) / 128, (3 * C_) / 128);
        qkv_gemm_kernel<<<grid, 128, 0, stream>>>(xbf, wqbf, b_qkv, qb, kb, vt);
    }
    // 4. RoPE on q and k (tokens 1..576)
    {
        int n = B_ * H_ * (N_ - 1) * (D_ / 2);
        rope_kernel<<<(n + 255) / 256, 256, 0, stream>>>(qb, kb);
    }
    // 5. flash attention: grid (19 query tiles of 32, 256 head-batches), 1 wave each
    {
        dim3 grid((N_ + 31) / 32, B_ * H_);
        attn_kernel<<<grid, 32, 0, stream>>>(qb, kb, vt, att);
    }
    // 6. LayerNorm
    ln_kernel<<<M_, 256, 0, stream>>>(att, ln_g, ln_b, xln);
    // 7. output projection (M=9232, N=1024, K=1024), 128x128 block tiles
    {
        dim3 grid((M_ + 127) / 128, C_ / 128);
        proj_gemm_kernel<<<grid, 128, 0, stream>>>(xln, wpbf, b_proj, out);
    }
}
